// LocalAggregator_84241488544065
// MI455X (gfx1250) — matrix operand order repaired and checked by reading, not compiled, not run on hardware
//
#include <hip/hip_runtime.h>
#include <hip/hip_bf16.h>

// MI455X (gfx1250) GAT-style local aggregator.
// One workgroup (256 thr = 8 wave32) per batch. Both GEMMs run on
// v_wmma_f32_16x16x32_bf16 (fp32 accumulate). h[b] is staged in LDS once
// (row-major + transposed bf16 copies); adj i-tiles are DMA'd into LDS by the
// Tensor Data Mover (tensor_load_to_lds / s_wait_tensorcnt), overlapped with
// the A-operand build. Roofline: 52 GFLOP total, ~187MB min HBM traffic
// (~8us @ 23.3TB/s), so the bf16 WMMA path keeps compute/memory balanced.

typedef __attribute__((ext_vector_type(16))) __bf16 v16bf;
typedef __attribute__((ext_vector_type(4)))  __bf16 v4bf;
typedef __attribute__((ext_vector_type(8)))  float  v8f;
typedef __attribute__((ext_vector_type(4)))  unsigned int v4u;
typedef __attribute__((ext_vector_type(8)))  int v8i;
typedef __attribute__((ext_vector_type(4)))  int v4i;

#define BATCH 512
#define NN    200
#define DD    256
#define NP    224          // N padded to 7 K-chunks of 32
#define HB_S  264          // Hb row stride (bf16)   -> 528B rows
#define HT_S  232          // HbT row stride (bf16)  -> 464B rows
#define PB_S  232          // Pb row stride (bf16)
#define NEG_INF_F (-9.0e15f)
#define SLOPE 0.2f

// LDS map (bytes):
//  Hb  [224][264] bf16  : 118272
//  HbT [256][232] bf16  : 118784
//  Ab  [4][16][264] bf16:  33792
//  Sb  [16][224] f32    :  14336
//  Pb  [16][232] bf16   :   7424
//  av  [4][256] f32     :   4096
//  Adjb[16][200] i32    :  12800   => total 309504 (< 320KB/WGP)
#define SMEM_BYTES 309504

union FragB { v16bf v; uint4 q[2]; };

#define WMMA_BF16(A, Bm, C) \
  __builtin_amdgcn_wmma_f32_16x16x32_bf16(false, (A), false, (Bm), (short)0, (C), false, false)

__device__ __forceinline__ v8f zero8() {
  v8f z;
#pragma unroll
  for (int i = 0; i < 8; ++i) z[i] = 0.0f;
  return z;
}

__global__ __launch_bounds__(256, 1)
void LocalAggregator_84241488544065_kernel(const float* __restrict__ hidden,
                                           const int*   __restrict__ adj,
                                           const float* __restrict__ a0,
                                           const float* __restrict__ a1,
                                           const float* __restrict__ a2,
                                           const float* __restrict__ a3,
                                           float*       __restrict__ out) {
  extern __shared__ __align__(16) char smem[];
  __bf16* Hb  = (__bf16*)smem;                 // [224][HB_S]
  __bf16* HbT = Hb  + 224 * HB_S;              // [256][HT_S]
  __bf16* Ab  = HbT + 256 * HT_S;              // [4][16][HB_S]
  float*  Sb  = (float*)(Ab + 4 * 16 * HB_S);  // [16][NP]
  __bf16* Pb  = (__bf16*)(Sb + 16 * NP);       // [16][PB_S]
  float*  av  = (float*)(Pb + 16 * PB_S);      // [4][DD]
  int*    Adjb = (int*)(av + 4 * DD);          // [16][NN]

  const int b    = blockIdx.x;
  const int tid  = threadIdx.x;
  const int lane = tid & 31;
  const int wid  = tid >> 5;
  const int hi   = (lane >> 4) & 1;
  const int l15  = lane & 15;

  const float* Hg = hidden + (size_t)b * NN * DD;

  // ---------- phase 0: stage per-batch data ----------
  for (int idx = tid; idx < 4 * DD; idx += 256) {
    const float* ap = (idx < DD) ? a0 : (idx < 2 * DD) ? a1 : (idx < 3 * DD) ? a2 : a3;
    av[idx] = ap[idx & (DD - 1)];
  }
  for (int idx = tid; idx < NN * (DD / 4); idx += 256) {
    const int j  = idx >> 6;             // DD/4 == 64 chunks per row
    const int dq = (idx & 63) << 2;
    const float4 h4 = *(const float4*)(Hg + j * DD + dq);
    v4bf v;
    v[0] = (__bf16)h4.x; v[1] = (__bf16)h4.y; v[2] = (__bf16)h4.z; v[3] = (__bf16)h4.w;
    *(v4bf*)(Hb + j * HB_S + dq) = v;          // row-major copy
    HbT[(dq + 0) * HT_S + j] = v[0];           // transposed copy
    HbT[(dq + 1) * HT_S + j] = v[1];
    HbT[(dq + 2) * HT_S + j] = v[2];
    HbT[(dq + 3) * HT_S + j] = v[3];
  }
  // zero pads: Hb rows 200..223, HbT cols 200..231, Sb cols 208..223 = -inf
  for (int idx = tid; idx < 24 * HB_S; idx += 256) Hb[200 * HB_S + idx] = (__bf16)0.0f;
  for (int idx = tid; idx < 256 * 32; idx += 256) {
    const int d = idx >> 5, j = 200 + (idx & 31);
    HbT[d * HT_S + j] = (__bf16)0.0f;
  }
  { const int m = tid >> 4, c = tid & 15; Sb[m * NP + 208 + c] = NEG_INF_F; }
  __syncthreads();

  // ---------- loop over 13 i-tiles of 16 rows ----------
  for (int it = 0; it < 13; ++it) {
    const int i0 = it * 16;

    // Kick off TDM DMA of adj tile [16 x 200] i32 into LDS; overlaps with 1a.
    // tensor_dim1 = NN - i0 so the tail tile's OOB rows are zero-filled by HW.
    if (wid == 0) {
      const unsigned long long ga =
          (unsigned long long)(const void*)(adj + ((size_t)b * NN + i0) * NN);
      const unsigned int ldsa = (unsigned int)(unsigned long long)Adjb;
      const unsigned int rem  = (unsigned int)(NN - i0);
      v4u g0;
      g0[0] = 1u;                                             // count=1
      g0[1] = ldsa;                                           // lds_addr
      g0[2] = (unsigned int)(ga & 0xFFFFFFFFull);             // global_addr lo
      g0[3] = (unsigned int)((ga >> 32) & 0x01FFFFFFull) | 0x80000000u; // hi | type=2
      v8i g1;
      g1[0] = (int)(2u << 16);             // workgroup_mask=0, data_size=2 (4B)
      g1[1] = (int)((unsigned)NN << 16);   // tensor_dim0 low16
      g1[2] = (int)(rem << 16);            // tensor_dim0 hi16=0 | tensor_dim1 low16
      g1[3] = (int)((unsigned)NN << 16);   // tensor_dim1 hi16=0 | tile_dim0=200
      g1[4] = 16;                          // tile_dim1=16, tile_dim2=0
      g1[5] = NN;                          // tensor_dim0_stride low32 = 200
      g1[6] = 0;                           // stride0 hi | stride1 low
      g1[7] = 0;
      v4i gz = {0, 0, 0, 0};
#if __clang_major__ >= 23
      v8i gz8 = {0, 0, 0, 0, 0, 0, 0, 0};
      __builtin_amdgcn_tensor_load_to_lds(g0, g1, gz, gz, gz8, 0);
#else
      __builtin_amdgcn_tensor_load_to_lds(g0, g1, gz, gz, 0);
#endif
    }

    // 1a: Ab[k][m][d] = bf16( Hb[i0+m][d] * a_k[d] )  (rows>=200 already zero)
    for (int idx = tid; idx < 4 * 16 * (DD / 4); idx += 256) {
      const int k   = idx >> 10;
      const int rem = idx & 1023;
      const int m   = rem >> 6;
      const int dq  = (rem & 63) << 2;
      const v4bf h  = *(const v4bf*)(Hb + (i0 + m) * HB_S + dq);
      v4bf r;
      r[0] = (__bf16)((float)h[0] * av[k * DD + dq + 0]);
      r[1] = (__bf16)((float)h[1] * av[k * DD + dq + 1]);
      r[2] = (__bf16)((float)h[2] * av[k * DD + dq + 2]);
      r[3] = (__bf16)((float)h[3] * av[k * DD + dq + 3]);
      *(v4bf*)(Ab + (k * 16 + m) * HB_S + dq) = r;
    }
    if (wid == 0) __builtin_amdgcn_s_wait_tensorcnt(0);
    __syncthreads();

    // 1b: edge scores; each wave drives TWO j-tiles sharing the 4 A-fragments
    // (8 WMMAs per 12 ds_load_b128 instead of 4 per 10).
    {
      const bool dual = (wid < 5);
      const int j0a = wid * 16;
      const int j0b = (dual ? (wid + 8) : wid) * 16;   // clamp keeps loads in range
      v8f aA0 = zero8(), aA1 = zero8(), aA2 = zero8(), aA3 = zero8();
      v8f aB0 = zero8(), aB1 = zero8(), aB2 = zero8(), aB3 = zero8();
      const __bf16* browA = Hb + (j0a + l15) * HB_S + hi * 8;
      const __bf16* browB = Hb + (j0b + l15) * HB_S + hi * 8;
      const __bf16* arow  = Ab + l15 * HB_S + hi * 8;          // +k*16*HB_S per k
#pragma unroll
      for (int c = 0; c < 8; ++c) {
        FragB bA, bB;
        bA.q[0] = *(const uint4*)(browA + c * 32);
        bA.q[1] = *(const uint4*)(browA + c * 32 + 16);
        bB.q[0] = *(const uint4*)(browB + c * 32);
        bB.q[1] = *(const uint4*)(browB + c * 32 + 16);
        FragB fa[2];
        fa[0].q[0] = *(const uint4*)(arow + c * 32);
        fa[0].q[1] = *(const uint4*)(arow + c * 32 + 16);
        fa[1].q[0] = *(const uint4*)(arow + 16 * HB_S + c * 32);
        fa[1].q[1] = *(const uint4*)(arow + 16 * HB_S + c * 32 + 16);
        aA0 = WMMA_BF16(fa[0].v, bA.v, aA0);
        aB0 = WMMA_BF16(fa[0].v, bB.v, aB0);
        fa[0].q[0] = *(const uint4*)(arow + 32 * HB_S + c * 32);
        fa[0].q[1] = *(const uint4*)(arow + 32 * HB_S + c * 32 + 16);
        aA1 = WMMA_BF16(fa[1].v, bA.v, aA1);
        aB1 = WMMA_BF16(fa[1].v, bB.v, aB1);
        fa[1].q[0] = *(const uint4*)(arow + 48 * HB_S + c * 32);
        fa[1].q[1] = *(const uint4*)(arow + 48 * HB_S + c * 32 + 16);
        aA2 = WMMA_BF16(fa[0].v, bA.v, aA2);
        aB2 = WMMA_BF16(fa[0].v, bB.v, aB2);
        aA3 = WMMA_BF16(fa[1].v, bA.v, aA3);
        aB3 = WMMA_BF16(fa[1].v, bB.v, aB3);
      }
      // leaky-relu + adj select; C layout: VGPR v -> M = v + 8*hi, lane -> N
      auto post = [&](int j0, const v8f& e0v, const v8f& e1v, const v8f& e2v,
                      const v8f& e3v) {
        const int gj = j0 + l15;
#pragma unroll
        for (int v = 0; v < 8; ++v) {
          const int m = v + hi * 8;
          float e0 = e0v[v]; e0 = e0 > 0.f ? e0 : SLOPE * e0;
          float e1 = e1v[v]; e1 = e1 > 0.f ? e1 : SLOPE * e1;
          float e2 = e2v[v]; e2 = e2 > 0.f ? e2 : SLOPE * e2;
          float e3 = e3v[v]; e3 = e3 > 0.f ? e3 : SLOPE * e3;
          float sel = NEG_INF_F;
          if (gj < NN) {
            const int a = Adjb[m * NN + gj];   // OOB rows DMA'd as 0 -> masked
            sel = (a == 1) ? e0 : (a == 2) ? e1 : (a == 3) ? e2 : (a == 4) ? e3
                                                                           : NEG_INF_F;
          }
          Sb[m * NP + j0 + l15] = sel;
        }
      };
      post(j0a, aA0, aA1, aA2, aA3);
      if (dual) post(j0b, aB0, aB1, aB2, aB3);
    }
    __syncthreads();

    // 1c: softmax over j (224 cols, padded cols are -inf -> exp underflows to 0)
#pragma unroll
    for (int r = 0; r < 2; ++r) {
      const int m = wid * 2 + r;
      const float* row = Sb + m * NP;
      float x[7];
      float mx = NEG_INF_F;
#pragma unroll
      for (int c = 0; c < 7; ++c) { x[c] = row[lane + 32 * c]; mx = fmaxf(mx, x[c]); }
#pragma unroll
      for (int s = 16; s > 0; s >>= 1) mx = fmaxf(mx, __shfl_xor(mx, s, 32));
      float sum = 0.f;
#pragma unroll
      for (int c = 0; c < 7; ++c) { x[c] = __expf(x[c] - mx); sum += x[c]; }
#pragma unroll
      for (int s = 16; s > 0; s >>= 1) sum += __shfl_xor(sum, s, 32);
      const float inv = 1.0f / sum;
#pragma unroll
      for (int c = 0; c < 7; ++c) Pb[m * PB_S + lane + 32 * c] = (__bf16)(x[c] * inv);
    }
    __syncthreads();

    // 1d: out[i-tile] = P[16x224] @ H[224x256]; each wave drives TWO d-tiles
    // sharing the P (A) fragment.
    {
      const int n0a = wid * 16, n0b = (wid + 8) * 16;
      v8f accA = zero8(), accB = zero8();
      const __bf16* prow  = Pb  + l15 * PB_S + hi * 8;
      const __bf16* trowA = HbT + (n0a + l15) * HT_S + hi * 8;
      const __bf16* trowB = HbT + (n0b + l15) * HT_S + hi * 8;
#pragma unroll
      for (int jc = 0; jc < 7; ++jc) {
        FragB fa, fbA, fbB;
        fa.q[0]  = *(const uint4*)(prow + jc * 32);
        fa.q[1]  = *(const uint4*)(prow + jc * 32 + 16);
        fbA.q[0] = *(const uint4*)(trowA + jc * 32);
        fbA.q[1] = *(const uint4*)(trowA + jc * 32 + 16);
        fbB.q[0] = *(const uint4*)(trowB + jc * 32);
        fbB.q[1] = *(const uint4*)(trowB + jc * 32 + 16);
        accA = WMMA_BF16(fa.v, fbA.v, accA);
        accB = WMMA_BF16(fa.v, fbB.v, accB);
      }
#pragma unroll
      for (int v = 0; v < 8; ++v) {
        const int gi = i0 + v + hi * 8;
        if (gi < NN) {
          out[((size_t)b * NN + gi) * DD + n0a + l15] = accA[v];
          out[((size_t)b * NN + gi) * DD + n0b + l15] = accB[v];
        }
      }
    }
    __syncthreads();
  }
}

extern "C" void kernel_launch(void* const* d_in, const int* in_sizes, int n_in,
                              void* d_out, int out_size, void* d_ws, size_t ws_size,
                              hipStream_t stream) {
  (void)in_sizes; (void)n_in; (void)d_ws; (void)ws_size; (void)out_size;
  const float* hidden = (const float*)d_in[0];
  const int*   adj    = (const int*)d_in[1];
  const float* a0     = (const float*)d_in[2];
  const float* a1     = (const float*)d_in[3];
  const float* a2     = (const float*)d_in[4];
  const float* a3     = (const float*)d_in[5];
  float*       outp   = (float*)d_out;

  // Opt into >64KB dynamic LDS (CDNA5 WGP allows up to 320KB per workgroup).
  (void)hipFuncSetAttribute((const void*)LocalAggregator_84241488544065_kernel,
                            hipFuncAttributeMaxDynamicSharedMemorySize, SMEM_BYTES);

  LocalAggregator_84241488544065_kernel<<<BATCH, 256, SMEM_BYTES, stream>>>(
      hidden, adj, a0, a1, a2, a3, outp);
}
